// Tier2MoE_24206435680284
// MI455X (gfx1250) — matrix operand order repaired and checked by reading, not compiled, not run on hardware
//
#include <hip/hip_runtime.h>

// ---------------- problem constants ----------------
#define B_  8
#define N_  2048
#define D_  512
#define C_  32
#define H_  1024
#define E_  8
#define T_  (B_*N_)          // 16384 tokens
#define MT  32               // tokens per block
#define HC  128              // H chunk size
#define LN_EPS 1e-5f

typedef __attribute__((ext_vector_type(16))) __bf16 v16bf;
typedef __attribute__((ext_vector_type(8)))  __bf16 v8bf;
typedef __attribute__((ext_vector_type(8)))  float  v8f;

__device__ __forceinline__ v8f v8f_zero() {
  v8f t = {0.f,0.f,0.f,0.f,0.f,0.f,0.f,0.f};
  return t;
}

// A-fragment as this code feeds v_wmma_*_16x16x32_bf16 (contiguous k order; B is packed to match):
// lanes 0-15 hold K[0..15] of row M=lane;
// lanes 16-31 hold K[16..31] of row M=lane-16.
__device__ __forceinline__ v16bf load_a16(const __bf16* p) {
  v8bf lo = *(const v8bf*)p;
  v8bf hi = *(const v8bf*)(p + 8);
  return __builtin_shufflevector(lo, hi, 0,1,2,3,4,5,6,7,8,9,10,11,12,13,14,15);
}

__device__ __forceinline__ v8f wmma_bf16(v16bf a, v16bf b, v8f c) {
  return __builtin_amdgcn_wmma_f32_16x16x32_bf16(false, a, false, b, (short)0, c, false, false);
}

// ---------------- kernel 0: pack f32 (K x Nc) row-major weights ----------------
// -> bf16 blocks dst[((e*(K/32)+kb)*Nc + n)*32 + j], j = K offset within block.
// A lane's B-fragment is then 16 contiguous bf16 (one aligned 32B v16bf load).
__global__ void pack_w_k(const float* __restrict__ src, __bf16* __restrict__ dst,
                         int K, int Nc) {
  int i = blockIdx.x * blockDim.x + threadIdx.x;
  int total = E_ * (K / 32) * Nc;
  if (i >= total) return;
  int n  = i % Nc;
  int kb = (i / Nc) % (K / 32);
  int e  = i / (Nc * (K / 32));
  const float* s = src + ((size_t)e * K + kb * 32) * Nc + n;
  __bf16* d = dst + ((size_t)i << 5);
  #pragma unroll
  for (int j = 0; j < 32; ++j) d[j] = (__bf16)s[(size_t)j * Nc];
}

// ---------------- kernel 1: router (softmax + top-2 combine weights) -------
__global__ void router_k(const float* __restrict__ z,
                         const float* __restrict__ rw,
                         const float* __restrict__ rb,
                         float* __restrict__ probs_out,   // (B,N,E) second tuple output
                         float* __restrict__ wcomb)       // (T,E) combine weights
{
  int gtid = blockIdx.x * blockDim.x + threadIdx.x;
  int t    = gtid >> 5;
  int lane = gtid & 31;
  if (t >= T_) return;
  const float* zp = z + (size_t)t * D_;
  float l[E_];
  #pragma unroll
  for (int e = 0; e < E_; ++e) {
    float s = 0.f;
    for (int k = lane; k < D_; k += 32) s += zp[k] * rw[e * D_ + k];
    #pragma unroll
    for (int off = 16; off > 0; off >>= 1) s += __shfl_down(s, off, 32);
    l[e] = s + rb[e];
  }
  if (lane == 0) {
    float mx = l[0];
    #pragma unroll
    for (int e = 1; e < E_; ++e) mx = fmaxf(mx, l[e]);
    float p[E_]; float sum = 0.f;
    #pragma unroll
    for (int e = 0; e < E_; ++e) { p[e] = __expf(l[e] - mx); sum += p[e]; }
    float inv = 1.f / sum;
    #pragma unroll
    for (int e = 0; e < E_; ++e) { p[e] *= inv; probs_out[(size_t)t * E_ + e] = p[e]; }
    // top-2 (lowest index wins ties, like jax top_k)
    int i1 = 0;
    #pragma unroll
    for (int e = 1; e < E_; ++e) if (p[e] > p[i1]) i1 = e;
    int i2 = (i1 == 0) ? 1 : 0;
    #pragma unroll
    for (int e = 0; e < E_; ++e) if (e != i1 && p[e] > p[i2]) i2 = e;
    float ws = p[i1] + p[i2] + 1e-9f;
    #pragma unroll
    for (int e = 0; e < E_; ++e)
      wcomb[(size_t)t * E_ + e] = (e == i1) ? p[i1] / ws : ((e == i2) ? p[i2] / ws : 0.f);
  }
}

// ---------------- kernel 2: cterm[b][e][h] = b1[e][h] + sum_c c_c[b][c]*wc[e][c][h]
__global__ void cterm_k(const float* __restrict__ c_c,
                        const float* __restrict__ wc,
                        const float* __restrict__ b1,
                        float* __restrict__ cterm)
{
  int i = blockIdx.x * blockDim.x + threadIdx.x;
  if (i >= B_ * E_ * H_) return;
  int h = i % H_;
  int e = (i / H_) % E_;
  int b = i / (E_ * H_);
  float s = b1[e * H_ + h];
  #pragma unroll 8
  for (int c = 0; c < C_; ++c) s += c_c[b * C_ + c] * wc[((size_t)e * C_ + c) * H_ + h];
  cterm[i] = s;
}

// ---------------- kernel 3: fused expert pipeline ----------------
__launch_bounds__(256, 1)
__global__ void moe_main_k(const float* __restrict__ z,
                           const __bf16* __restrict__ w1p,   // packed bf16
                           const __bf16* __restrict__ w2p,
                           const __bf16* __restrict__ wvp,
                           const __bf16* __restrict__ wgp,
                           const float* __restrict__ b2,
                           const float* __restrict__ bvb,
                           const float* __restrict__ bgb,
                           const float* __restrict__ gamma,
                           const float* __restrict__ beta,
                           const float* __restrict__ cterm,
                           const float* __restrict__ wcomb,
                           float* __restrict__ out)
{
  // 32 KB: z tile in bf16, persistent across the expert loop
  __shared__ __align__(16) __bf16 zt[MT * D_];
  // 32 KB scratch with disjoint lifetimes: h-chunk (bf16), then h2 (bf16), then LN stats
  __shared__ __align__(16) unsigned char sbuf[MT * D_ * 2];
  __bf16* hch    = (__bf16*)sbuf;   // [MT][HC]
  __bf16* h2t    = (__bf16*)sbuf;   // [MT][D_]
  float*  rsum   = (float*)sbuf;    // [MT]
  float*  rsumsq = ((float*)sbuf) + MT;

  const int tid    = threadIdx.x;
  const int wid    = tid >> 5;          // wave id 0..7
  const int lane   = tid & 31;
  const int lo16   = lane & 15;
  const int hihalf = (lane >= 16) ? 1 : 0;
  const int akoff  = hihalf ? 16 : 0;   // A-frag K sub-offset (elements)
  const int bkoff  = hihalf ? 16 : 0;   // B-frag K sub-offset (elements)
  const int t0     = blockIdx.x * MT;
  const int bIdx   = t0 / N_;           // MT divides N_, tiles never straddle batches

  // stage z tile -> bf16 LDS (contiguous, coalesced)
  for (int i = tid; i < MT * D_; i += 256)
    zt[i] = (__bf16)z[(size_t)t0 * D_ + i];

  // per-lane output accumulator: 2 M-tiles x 4 N-tiles of 16x16 (cols wid*64..wid*64+63)
  v8f oaccA[2][4];
  #pragma unroll
  for (int mt = 0; mt < 2; ++mt)
    #pragma unroll
    for (int nt = 0; nt < 4; ++nt) oaccA[mt][nt] = v8f_zero();

  __syncthreads();

  for (int e = 0; e < E_; ++e) {
    const __bf16* w1e = w1p + (size_t)e * (D_ / 32) * H_ * 32;
    const __bf16* w2e = w2p + (size_t)e * (H_ / 32) * D_ * 32;
    const __bf16* wve = wvp + (size_t)e * (D_ / 32) * D_ * 32;
    const __bf16* wge = wgp + (size_t)e * (D_ / 32) * D_ * 32;

    // h2 accumulator tiles (32 x 512 spread over 8 waves)
    v8f acc2[2][4];
    #pragma unroll
    for (int mt = 0; mt < 2; ++mt)
      #pragma unroll
      for (int nt = 0; nt < 4; ++nt) acc2[mt][nt] = v8f_zero();

    // ---- stages 1+2, chunked over H ----
    for (int hc = 0; hc < H_ / HC; ++hc) {
      // stage 1: h_chunk = ELU(z @ w1[:, chunk] + cterm); wave wid owns N-tile wid
      v8f h1[2] = { v8f_zero(), v8f_zero() };
      const int colH = hc * HC + wid * 16 + lo16;
      #pragma unroll 2
      for (int ks = 0; ks < D_ / 32; ++ks) {
        const int k0 = ks * 32;
        const __bf16* wp = w1e + ((size_t)ks * H_ + colH) * 32 + bkoff;
        __builtin_prefetch(wp + (size_t)H_ * 32, 0, 0);
        v16bf bfrag = *(const v16bf*)wp;             // 32B aligned, coalesced
        #pragma unroll
        for (int mt = 0; mt < 2; ++mt) {
          v16bf af = load_a16(&zt[(mt * 16 + lo16) * D_ + k0 + akoff]);
          h1[mt] = wmma_bf16(af, bfrag, h1[mt]);
        }
      }
      const float ct = cterm[((size_t)bIdx * E_ + e) * H_ + colH];
      #pragma unroll
      for (int mt = 0; mt < 2; ++mt)
        #pragma unroll
        for (int r = 0; r < 8; ++r) {
          float hv = h1[mt][r] + ct;
          hv = hv > 0.f ? hv : (__expf(hv) - 1.f);       // ELU
          hch[(mt * 16 + hihalf * 8 + r) * HC + wid * 16 + lo16] = (__bf16)hv;
        }
      __syncthreads();

      // stage 2: acc2 += h_chunk @ w2[chunk, :]; wave wid owns cols wid*64..+63
      for (int ks = 0; ks < HC / 32; ++ks) {
        const int k0 = ks * 32;
        const int kb = hc * (HC / 32) + ks;            // global K-block in H
        v16bf af[2];
        #pragma unroll
        for (int mt = 0; mt < 2; ++mt)
          af[mt] = load_a16(&hch[(mt * 16 + lo16) * HC + k0 + akoff]);
        #pragma unroll
        for (int nt = 0; nt < 4; ++nt) {
          const int colD = wid * 64 + nt * 16 + lo16;
          v16bf bf2 = *(const v16bf*)(w2e + ((size_t)kb * D_ + colD) * 32 + bkoff);
          acc2[0][nt] = wmma_bf16(af[0], bf2, acc2[0][nt]);
          acc2[1][nt] = wmma_bf16(af[1], bf2, acc2[1][nt]);
        }
      }
      __syncthreads();   // hch reused next chunk
    }

    // h2 = acc2 + b2 -> bf16 LDS (A-panel for stage 3)
    #pragma unroll
    for (int mt = 0; mt < 2; ++mt)
      #pragma unroll
      for (int nt = 0; nt < 4; ++nt) {
        const int colD = wid * 64 + nt * 16 + lo16;
        const float bb = b2[(size_t)e * D_ + colD];
        #pragma unroll
        for (int r = 0; r < 8; ++r)
          h2t[(mt * 16 + hihalf * 8 + r) * D_ + colD] = (__bf16)(acc2[mt][nt][r] + bb);
      }
    __syncthreads();

    // ---- stage 3: v = h2@wv, g = h2@wg, GLU; per N-tile to bound registers ----
    v8f yv[2][4];
    #pragma unroll
    for (int nt = 0; nt < 4; ++nt) {
      const int colD = wid * 64 + nt * 16 + lo16;
      v8f vac[2] = { v8f_zero(), v8f_zero() };
      v8f gac[2] = { v8f_zero(), v8f_zero() };
      #pragma unroll 2
      for (int ks = 0; ks < D_ / 32; ++ks) {
        const int k0 = ks * 32;
        v16bf af[2];
        #pragma unroll
        for (int mt = 0; mt < 2; ++mt)
          af[mt] = load_a16(&h2t[(mt * 16 + lo16) * D_ + k0 + akoff]);
        const __bf16* wpv = wve + ((size_t)ks * D_ + colD) * 32 + bkoff;
        const __bf16* wpg = wge + ((size_t)ks * D_ + colD) * 32 + bkoff;
        __builtin_prefetch(wpv + (size_t)D_ * 32, 0, 0);
        v16bf bfv = *(const v16bf*)wpv;
        v16bf bfg = *(const v16bf*)wpg;
        vac[0] = wmma_bf16(af[0], bfv, vac[0]);
        vac[1] = wmma_bf16(af[1], bfv, vac[1]);
        gac[0] = wmma_bf16(af[0], bfg, gac[0]);
        gac[1] = wmma_bf16(af[1], bfg, gac[1]);
      }
      const float bvc = bvb[(size_t)e * D_ + colD];
      const float bgc = bgb[(size_t)e * D_ + colD];
      #pragma unroll
      for (int mt = 0; mt < 2; ++mt)
        #pragma unroll
        for (int r = 0; r < 8; ++r) {
          float vv = vac[mt][r] + bvc;
          float gg = gac[mt][r] + bgc;
          yv[mt][nt][r] = vv * (1.f / (1.f + __expf(-gg)));   // GLU
        }
    }
    __syncthreads();   // h2t dead; sbuf becomes LN row stats

    if (tid < 2 * MT) rsum[tid] = 0.f;     // zeros rsum[0..31] and rsumsq[0..31]
    __syncthreads();

    // residual + per-row partial sums
    float psum[2][8] = {}, psq[2][8] = {};
    #pragma unroll
    for (int mt = 0; mt < 2; ++mt)
      #pragma unroll
      for (int nt = 0; nt < 4; ++nt) {
        const int colD = wid * 64 + nt * 16 + lo16;
        #pragma unroll
        for (int r = 0; r < 8; ++r) {
          const int m = mt * 16 + hihalf * 8 + r;
          float y = yv[mt][nt][r] + z[(size_t)(t0 + m) * D_ + colD];
          yv[mt][nt][r] = y;
          psum[mt][r] += y;
          psq[mt][r]  += y * y;
        }
      }
    #pragma unroll
    for (int mt = 0; mt < 2; ++mt)
      #pragma unroll
      for (int r = 0; r < 8; ++r) {
        const int m = mt * 16 + hihalf * 8 + r;
        atomicAdd(&rsum[m],   psum[mt][r]);
        atomicAdd(&rsumsq[m], psq[mt][r]);
      }
    __syncthreads();

    // layernorm + gamma/beta + weighted combine into output accumulator
    #pragma unroll
    for (int mt = 0; mt < 2; ++mt)
      #pragma unroll
      for (int r = 0; r < 8; ++r) {
        const int m   = mt * 16 + hihalf * 8 + r;
        const float mu   = rsum[m] * (1.f / D_);
        const float var  = rsumsq[m] * (1.f / D_) - mu * mu;
        const float rstd = rsqrtf(var + LN_EPS);
        const float wgt  = wcomb[(size_t)(t0 + m) * E_ + e];
        #pragma unroll
        for (int nt = 0; nt < 4; ++nt) {
          const int colD = wid * 64 + nt * 16 + lo16;
          float yn = (yv[mt][nt][r] - mu) * rstd * gamma[(size_t)e * D_ + colD]
                     + beta[(size_t)e * D_ + colD];
          oaccA[mt][nt][r] += wgt * yn;
        }
      }
    __syncthreads();   // sbuf reused as hch by next expert
  }

  // write output tile
  #pragma unroll
  for (int mt = 0; mt < 2; ++mt)
    #pragma unroll
    for (int nt = 0; nt < 4; ++nt) {
      const int colD = wid * 64 + nt * 16 + lo16;
      #pragma unroll
      for (int r = 0; r < 8; ++r) {
        const int m = mt * 16 + hihalf * 8 + r;
        out[(size_t)(t0 + m) * D_ + colD] = oaccA[mt][nt][r];
      }
    }
}

// ---------------- launch ----------------
extern "C" void kernel_launch(void* const* d_in, const int* in_sizes, int n_in,
                              void* d_out, int out_size, void* d_ws, size_t ws_size,
                              hipStream_t stream) {
  (void)in_sizes; (void)n_in; (void)out_size; (void)ws_size;
  const float* z     = (const float*)d_in[0];
  const float* c_c   = (const float*)d_in[1];
  const float* rw    = (const float*)d_in[2];
  const float* rb    = (const float*)d_in[3];
  const float* w1    = (const float*)d_in[4];
  const float* b1    = (const float*)d_in[5];
  const float* wc    = (const float*)d_in[6];
  const float* w2    = (const float*)d_in[7];
  const float* b2    = (const float*)d_in[8];
  const float* wvf   = (const float*)d_in[9];
  const float* bvp   = (const float*)d_in[10];
  const float* wgf   = (const float*)d_in[11];
  const float* bgp   = (const float*)d_in[12];
  const float* gamma = (const float*)d_in[13];
  const float* beta  = (const float*)d_in[14];

  float* out   = (float*)d_out;                    // (B,N,D)
  float* probs = out + (size_t)B_ * N_ * D_;       // (B,N,E) — second tuple output

  // workspace layout
  float*  wcomb = (float*)d_ws;                    // (T,E)         512 KB
  float*  cterm = wcomb + (size_t)T_ * E_;         // (B,E,H)       256 KB
  __bf16* w1p   = (__bf16*)(cterm + (size_t)B_ * E_ * H_);     // 8 MB
  __bf16* w2p   = w1p + (size_t)E_ * D_ * H_;                  // 8 MB
  __bf16* wvpk  = w2p + (size_t)E_ * H_ * D_;                  // 4 MB
  __bf16* wgpk  = wvpk + (size_t)E_ * D_ * D_;                 // 4 MB

  // pack weights to bf16 K-major blocks (once per launch; L2-resident afterwards)
  pack_w_k<<<dim3((E_ * (D_/32) * H_ + 255) / 256), 256, 0, stream>>>(w1,  w1p,  D_, H_);
  pack_w_k<<<dim3((E_ * (H_/32) * D_ + 255) / 256), 256, 0, stream>>>(w2,  w2p,  H_, D_);
  pack_w_k<<<dim3((E_ * (D_/32) * D_ + 255) / 256), 256, 0, stream>>>(wvf, wvpk, D_, D_);
  pack_w_k<<<dim3((E_ * (D_/32) * D_ + 255) / 256), 256, 0, stream>>>(wgf, wgpk, D_, D_);

  router_k<<<dim3((T_ * 32) / 256), 256, 0, stream>>>(z, rw, rb, probs, wcomb);
  cterm_k<<<dim3((B_ * E_ * H_ + 255) / 256), 256, 0, stream>>>(c_c, wc, b1, cterm);
  moe_main_k<<<dim3(T_ / MT), 256, 0, stream>>>(z, w1p, w2p, wvpk, wgpk, b2, bvp, bgp,
                                                gamma, beta, cterm, wcomb, out);
}
